// Model_82188494176644
// MI455X (gfx1250) — matrix operand order repaired and checked by reading, not compiled, not run on hardware
//
#include <hip/hip_runtime.h>
#include <hip/hip_bf16.h>

typedef __bf16 bf16;
typedef __attribute__((ext_vector_type(16))) bf16 v16bf;
typedef __attribute__((ext_vector_type(8)))  float v8f;

#define NPTS   2048
#define KNN    16
#define NB     16
#define NJ     16
#define BN_EPS 1e-5f
#define TWO_PI 6.28318530717958647692f

// Activation layout ("tiled"): value(channel k, point n) lives at
//   buf[(n/16)*16*C + (n%16)*C + k]
// so a WMMA B-fragment (32 K at one column; each lane loads two 8-K runs) is contiguous.

// ---------------------------------------------------------------------------
// Param preprocessing
// ---------------------------------------------------------------------------
__global__ void k_convert_w(const float* __restrict__ src, bf16* __restrict__ dst, int n) {
    int i = blockIdx.x * 256 + threadIdx.x;
    if (i < n) dst[i] = (bf16)src[i];
}

__global__ void k_fold_bn(const float* __restrict__ g, const float* __restrict__ b,
                          const float* __restrict__ m, const float* __restrict__ v,
                          float* __restrict__ a, float* __restrict__ c, int n) {
    int i = blockIdx.x * 256 + threadIdx.x;
    if (i < n) {
        float s = g[i] * rsqrtf(v[i] + BN_EPS);
        a[i] = s;
        c[i] = b[i] - m[i] * s;
    }
}

__global__ void k_fill_head(const float* __restrict__ hb, float* __restrict__ a, float* __restrict__ c) {
    int i = threadIdx.x;
    if (i < NJ) { a[i] = 1.0f; c[i] = hb[i]; }
}

// ---------------------------------------------------------------------------
// kNN: keep top-(K+1) of pd = 2<q,c> - |q|^2 - |c|^2 (largest == nearest),
// drop entry 0 (self). Candidate tiles staged to LDS with the CDNA5 async
// global->LDS copy path (ASYNCcnt), |c|^2 computed from LDS on the fly.
// ---------------------------------------------------------------------------
__global__ void k_knn(const float* __restrict__ pts, int* __restrict__ nidx) {
    __shared__ float tile[256][3];
    const int n = blockIdx.x * 256 + threadIdx.x;
    const float qx = pts[n * 3 + 0], qy = pts[n * 3 + 1], qz = pts[n * 3 + 2];
    const float qq = qx * qx + qy * qy + qz * qz;
    const float NEG_INF = -__builtin_inff();

    float tv[KNN + 1];
    int   ti[KNN + 1];
#pragma unroll
    for (int s = 0; s <= KNN; ++s) { tv[s] = NEG_INF; ti[s] = 0; }

    const unsigned int ldsoff =
        (unsigned int)(unsigned long long)(&tile[threadIdx.x][0]);  // LDS aperture: addr[31:0]

    for (int c0 = 0; c0 < NPTS; c0 += 256) {
        const float* src = pts + (size_t)(c0 + threadIdx.x) * 3;
        // async copy 12B/lane into LDS (offset applies to both LDS and global addr)
        asm volatile("global_load_async_to_lds_b32 %0, %1, off"
                     :: "v"(ldsoff), "v"(src) : "memory");
        asm volatile("global_load_async_to_lds_b32 %0, %1, off offset:4"
                     :: "v"(ldsoff), "v"(src) : "memory");
        asm volatile("global_load_async_to_lds_b32 %0, %1, off offset:8"
                     :: "v"(ldsoff), "v"(src) : "memory");
        asm volatile("s_wait_asynccnt 0" ::: "memory");
        __syncthreads();
        for (int j = 0; j < 256; ++j) {
            const float cx = tile[j][0], cy = tile[j][1], cz = tile[j][2];
            const float pd = 2.0f * (qx * cx + qy * cy + qz * cz) - qq
                           - (cx * cx + cy * cy + cz * cz);
            if (pd > tv[KNN]) {
                tv[KNN] = pd; ti[KNN] = c0 + j;
#pragma unroll
                for (int s = KNN; s > 0; --s) {
                    if (tv[s] > tv[s - 1]) {
                        float fv = tv[s]; tv[s] = tv[s - 1]; tv[s - 1] = fv;
                        int   iv = ti[s]; ti[s] = ti[s - 1]; ti[s - 1] = iv;
                    }
                }
            }
        }
        __syncthreads();
    }
#pragma unroll
    for (int i = 0; i < KNN; ++i) nidx[n * KNN + i] = ti[i + 1];
}

// ---------------------------------------------------------------------------
// RRI features -> tiled bf16 buffer with C=64 channels (channel = 4*i + t).
// 16 lanes cooperate per point (8 points / 128-thread block).
// ---------------------------------------------------------------------------
__global__ void k_rri(const float* __restrict__ pts, const int* __restrict__ nidx,
                      bf16* __restrict__ feat) {
    __shared__ float sT[8][KNN][3];
    __shared__ float sPn[8][3];
    const int lp = threadIdx.x >> 4;   // local point 0..7
    const int i  = threadIdx.x & 15;   // neighbor index
    const int n  = blockIdx.x * 8 + lp;

    const float px = pts[n * 3 + 0], py = pts[n * 3 + 1], pz = pts[n * 3 + 2];
    const float rp  = sqrtf(px * px + py * py + pz * pz);
    const float irp = 1.0f / rp;
    const float pnx = px * irp, pny = py * irp, pnz = pz * irp;

    const int qi = nidx[n * KNN + i];
    const float qx = pts[qi * 3 + 0], qy = pts[qi * 3 + 1], qz = pts[qi * 3 + 2];
    const float rq  = sqrtf(qx * qx + qy * qy + qz * qz);
    const float irq = 1.0f / rq;
    const float dot = (pnx * qx + pny * qy + pnz * qz) * irq;   // pn . qn
    const float th  = acosf(fminf(1.0f, fmaxf(-1.0f, dot)));
    const float Tx = qx - dot * px;                             // Tq = q - dot*p
    const float Ty = qy - dot * py;
    const float Tz = qz - dot * pz;

    sT[lp][i][0] = Tx; sT[lp][i][1] = Ty; sT[lp][i][2] = Tz;
    if (i == 0) { sPn[lp][0] = pnx; sPn[lp][1] = pny; sPn[lp][2] = pnz; }
    __syncthreads();

    const float bpx = sPn[lp][0], bpy = sPn[lp][1], bpz = sPn[lp][2];
    float m1 = __builtin_inff(), m2 = __builtin_inff();
#pragma unroll
    for (int j = 0; j < KNN; ++j) {
        const float ax = sT[lp][j][0], ay = sT[lp][j][1], az = sT[lp][j][2];
        // sin_psi = pn . (Tq_j x Tq_i)
        const float cxv = ay * Tz - az * Ty;
        const float cyv = az * Tx - ax * Tz;
        const float czv = ax * Ty - ay * Tx;
        const float s   = bpx * cxv + bpy * cyv + bpz * czv;
        const float cth = ax * Tx + ay * Ty + az * Tz;
        float ps = atan2f(s, cth);
        if (ps < 0.0f) ps += TWO_PI;                 // mod 2*pi
        if (ps < m1)      { m2 = m1; m1 = ps; }
        else if (ps < m2) { m2 = ps; }
    }
    const float phi = m2;   // row-sorted element 1 (psi[i][i]==0 is the min)

    union { bf16 h[4]; uint2 u; } pk;
    pk.h[0] = (bf16)rp; pk.h[1] = (bf16)rq; pk.h[2] = (bf16)th; pk.h[3] = (bf16)phi;
    // tiled write: C=64, channels 4i..4i+3 contiguous -> one 8B store
    *(uint2*)(feat + (size_t)(n >> 4) * (16 * 64) + (n & 15) * 64 + 4 * i) = pk.u;
}

// ---------------------------------------------------------------------------
// Fused GEMM + BN + ReLU:  Y = act( a * (W X) + c )
// W: COUT x CIN bf16 (row-major).  X: tiled bf16 (16 cols/panel, K contig).
// Block = 4 waves; every wave owns rows [bx*16,+16) x 4 column panels,
// sharing one A-fragment across 4 back-to-back WMMAs.
// Compile-time CIN/COUT + explicit ping-pong double buffering (step 64):
// no buffer copies, no pointer spills, partial loadcnt waits overlap the
// matrix pipe with memory. CIN must be a multiple of 32 (all layers are).
// ---------------------------------------------------------------------------
template <int CIN, int COUT, int RELU, int F32OUT>
__global__ void k_gemm_t(const bf16* __restrict__ Wb,
                         const float* __restrict__ av, const float* __restrict__ cv,
                         const bf16* __restrict__ X,
                         bf16* __restrict__ Yb, float* __restrict__ Yf) {
    const int lane  = threadIdx.x & 31;
    const int wave  = threadIdx.x >> 5;
    const int nn    = lane & 15;
    const int hi    = lane >> 4;                       // 0 or 1
    const int rowB  = blockIdx.x * 16;                 // 16 output rows per block
    const int pbase = blockIdx.y * 16 + wave * 4;      // 4 column panels per wave

    typedef union { uint4 u[2]; v16bf v; } frag_t;

    v8f acc[4] = {{}, {}, {}, {}};
    const bf16* wrow = Wb + (size_t)(rowB + nn) * CIN;
    const bf16* xcol[4];
#pragma unroll
    for (int p = 0; p < 4; ++p)
        xcol[p] = X + ((size_t)(pbase + p) * 16 + nn) * CIN;

    frag_t fa[2], fb[2][4];

    auto loadStage = [&](int buf, int k) {
        fa[buf].u[0] = *(const uint4*)(wrow + k + hi * 8);
        fa[buf].u[1] = *(const uint4*)(wrow + k + 16 + hi * 8);
#pragma unroll
        for (int p = 0; p < 4; ++p) {
            fb[buf][p].u[0] = *(const uint4*)(xcol[p] + k + hi * 8);
            fb[buf][p].u[1] = *(const uint4*)(xcol[p] + k + 16 + hi * 8);
        }
    };
    auto mma = [&](int buf) {
#pragma unroll
        for (int p = 0; p < 4; ++p)
            acc[p] = __builtin_amdgcn_wmma_f32_16x16x32_bf16(
                false, fa[buf].v, false, fb[buf][p].v, (short)0, acc[p], false, false);
    };

    loadStage(0, 0);
    for (int k0 = 0; k0 < CIN; k0 += 64) {
        if (k0 + 32 < CIN) {
            loadStage(1, k0 + 32);
            __builtin_prefetch((const void*)(xcol[0] + k0 + 64), 0, 1);
        }
        mma(0);
        if (k0 + 64 < CIN) loadStage(0, k0 + 64);
        if (k0 + 32 < CIN) mma(1);
    }

    // epilogue: C layout -> lane holds 8 consecutive K (rows) at one column
#pragma unroll
    for (int p = 0; p < 4; ++p) {
        float y[8];
#pragma unroll
        for (int e = 0; e < 8; ++e) {
            const int r = rowB + hi * 8 + e;
            float t = av[r] * acc[p][e] + cv[r];
            if (RELU) t = fmaxf(t, 0.0f);
            y[e] = t;
        }
        if (F32OUT) {   // head: logits stored [n][j] contiguous
            const int col = (pbase + p) * 16 + nn;
            float4* dst = (float4*)(Yf + (size_t)col * NJ + hi * 8);
            dst[0] = make_float4(y[0], y[1], y[2], y[3]);
            dst[1] = make_float4(y[4], y[5], y[6], y[7]);
        } else {        // tiled bf16: 8 consecutive K -> one 16B store
            union { bf16 h[8]; uint4 u; } o;
#pragma unroll
            for (int e = 0; e < 8; ++e) o.h[e] = (bf16)y[e];
            *(uint4*)(Yb + (size_t)(pbase + p) * 16 * COUT
                          + (size_t)nn * COUT + rowB + hi * 8) = o.u;
        }
    }
}

// ---------------------------------------------------------------------------
// Global max-pool per channel over tiled F (C=1024), then concat
// ---------------------------------------------------------------------------
__global__ void k_maxpool(const bf16* __restrict__ F, bf16* __restrict__ glob) {
    __shared__ float red[256];
    const int ch = blockIdx.x;
    float m = -__builtin_inff();
    for (int n = threadIdx.x; n < NPTS; n += 256)
        m = fmaxf(m, (float)F[(size_t)(n >> 4) * (16 * 1024) + (n & 15) * 1024 + ch]);
    red[threadIdx.x] = m;
    __syncthreads();
    for (int s = 128; s > 0; s >>= 1) {
        if (threadIdx.x < s) red[threadIdx.x] = fmaxf(red[threadIdx.x], red[threadIdx.x + s]);
        __syncthreads();
    }
    if (threadIdx.x == 0) glob[ch] = (bf16)red[0];
}

// output X tiled with C=2048: flat index i = panel*32768 + nn*2048 + c
__global__ void k_concat(const bf16* __restrict__ F, const bf16* __restrict__ glob,
                         bf16* __restrict__ X) {
    const int i = blockIdx.x * 256 + threadIdx.x;
    if (i >= 2048 * NPTS) return;
    const int c     = i & 2047;
    const int nn    = (i >> 11) & 15;
    const int panel = i >> 15;
    X[i] = (c < 1024) ? F[(size_t)panel * (16 * 1024) + nn * 1024 + c] : glob[c - 1024];
}

// ---------------------------------------------------------------------------
// Softmax over J=16 per point; logits [n][j] contiguous; gamma (J, NPTS)
// ---------------------------------------------------------------------------
__global__ void k_softmax(const float* __restrict__ logits, float* __restrict__ gamma) {
    const int n = blockIdx.x * 256 + threadIdx.x;
    float v[NJ];
    float mx = -__builtin_inff();
#pragma unroll
    for (int j = 0; j < NJ; ++j) { v[j] = logits[n * NJ + j]; mx = fmaxf(mx, v[j]); }
    float s = 0.0f;
#pragma unroll
    for (int j = 0; j < NJ; ++j) { v[j] = expf(v[j] - mx); s += v[j]; }
    const float is = 1.0f / s;
#pragma unroll
    for (int j = 0; j < NJ; ++j) gamma[j * NPTS + n] = v[j] * is;
}

// ---------------------------------------------------------------------------
// GMM parameters per (cloud,batch): pi, mu, isotropic sigma
// block = 16 clusters x 16 chunk-threads
// ---------------------------------------------------------------------------
__global__ void k_gmm(const float* __restrict__ gamma, const float* __restrict__ pts,
                      float* __restrict__ pi, float* __restrict__ mu, float* __restrict__ sig) {
    __shared__ float red[16][16][4];
    __shared__ float smu[16][3];
    __shared__ float sNpi[16];
    const int j = threadIdx.x & 15;
    const int t = threadIdx.x >> 4;

    float sg = 0, sx = 0, sy = 0, sz = 0;
    for (int n = t; n < NPTS; n += 16) {
        const float g = gamma[j * NPTS + n];
        sg += g;
        sx += g * pts[n * 3 + 0];
        sy += g * pts[n * 3 + 1];
        sz += g * pts[n * 3 + 2];
    }
    red[j][t][0] = sg; red[j][t][1] = sx; red[j][t][2] = sy; red[j][t][3] = sz;
    __syncthreads();
    if (t == 0) {
        float a0 = 0, a1 = 0, a2 = 0, a3 = 0;
        for (int u = 0; u < 16; ++u) {
            a0 += red[j][u][0]; a1 += red[j][u][1]; a2 += red[j][u][2]; a3 += red[j][u][3];
        }
        const float Npi = a0;
        const float inv = 1.0f / Npi;
        sNpi[j] = Npi;
        smu[j][0] = a1 * inv; smu[j][1] = a2 * inv; smu[j][2] = a3 * inv;
        pi[j] = Npi / (float)NPTS;
        mu[j * 3 + 0] = smu[j][0];
        mu[j * 3 + 1] = smu[j][1];
        mu[j * 3 + 2] = smu[j][2];
    }
    __syncthreads();
    const float mxv = smu[j][0], myv = smu[j][1], mzv = smu[j][2];
    float sd = 0;
    for (int n = t; n < NPTS; n += 16) {
        const float g  = gamma[j * NPTS + n];
        const float dx = pts[n * 3 + 0] - mxv;
        const float dy = pts[n * 3 + 1] - myv;
        const float dz = pts[n * 3 + 2] - mzv;
        sd += g * (dx * dx + dy * dy + dz * dz);
    }
    red[j][t][0] = sd;
    __syncthreads();
    if (t == 0) {
        float s = 0;
        for (int u = 0; u < 16; ++u) s += red[j][u][0];
        sig[j] = s / sNpi[j];
    }
}

// ---------------------------------------------------------------------------
// Registration: weighted covariance + 3x3 SVD (one-sided Jacobi) + Kabsch
// one thread per batch
// ---------------------------------------------------------------------------
__device__ inline float det3x3(const float M[3][3]) {
    return M[0][0] * (M[1][1] * M[2][2] - M[1][2] * M[2][1])
         - M[0][1] * (M[1][0] * M[2][2] - M[1][2] * M[2][0])
         + M[0][2] * (M[1][0] * M[2][1] - M[1][1] * M[2][0]);
}

__global__ void k_register(const float* __restrict__ pi1, const float* __restrict__ mu1,
                           const float* __restrict__ mu2, const float* __restrict__ sig2,
                           float* __restrict__ out) {
    const int b = threadIdx.x;
    if (b >= NB) return;
    const float* pib = pi1  + b * NJ;
    const float* m1  = mu1  + b * NJ * 3;
    const float* m2  = mu2  + b * NJ * 3;
    const float* s2  = sig2 + b * NJ;

    float cs[3] = {0, 0, 0}, ct[3] = {0, 0, 0};
    for (int j = 0; j < NJ; ++j) {
        const float p = pib[j];
        cs[0] += p * m1[j * 3 + 0]; cs[1] += p * m1[j * 3 + 1]; cs[2] += p * m1[j * 3 + 2];
        ct[0] += p * m2[j * 3 + 0]; ct[1] += p * m2[j * 3 + 1]; ct[2] += p * m2[j * 3 + 2];
    }
    float M[3][3] = {{0, 0, 0}, {0, 0, 0}, {0, 0, 0}};
    for (int j = 0; j < NJ; ++j) {
        const float w = pib[j] / s2[j];
        const float dsv[3] = {m1[j * 3] - cs[0], m1[j * 3 + 1] - cs[1], m1[j * 3 + 2] - cs[2]};
        const float dtv[3] = {m2[j * 3] - ct[0], m2[j * 3 + 1] - ct[1], m2[j * 3 + 2] - ct[2]};
        for (int a = 0; a < 3; ++a)
            for (int c = 0; c < 3; ++c)
                M[a][c] += w * dsv[a] * dtv[c];
    }

    // one-sided Jacobi: orthogonalize columns of G, accumulate V  (M = U S V^T)
    float G[3][3], V[3][3];
    for (int r = 0; r < 3; ++r)
        for (int c = 0; c < 3; ++c) { G[r][c] = M[r][c]; V[r][c] = (r == c) ? 1.0f : 0.0f; }
    const int PP[3] = {0, 0, 1}, QQ[3] = {1, 2, 2};
    for (int sweep = 0; sweep < 12; ++sweep) {
        for (int pq = 0; pq < 3; ++pq) {
            const int p = PP[pq], q = QQ[pq];
            float app = 0, aqq = 0, apq = 0;
            for (int r = 0; r < 3; ++r) {
                app += G[r][p] * G[r][p];
                aqq += G[r][q] * G[r][q];
                apq += G[r][p] * G[r][q];
            }
            if (fabsf(apq) > 1e-20f) {
                const float zeta = (aqq - app) / (2.0f * apq);
                const float tt = copysignf(1.0f, zeta) / (fabsf(zeta) + sqrtf(1.0f + zeta * zeta));
                const float c_ = rsqrtf(1.0f + tt * tt);
                const float s_ = c_ * tt;
                for (int r = 0; r < 3; ++r) {
                    float gp = G[r][p], gq = G[r][q];
                    G[r][p] = c_ * gp - s_ * gq; G[r][q] = s_ * gp + c_ * gq;
                    float vp = V[r][p], vq = V[r][q];
                    V[r][p] = c_ * vp - s_ * vq; V[r][q] = s_ * vp + c_ * vq;
                }
            }
        }
    }
    float sg[3];
    for (int c = 0; c < 3; ++c)
        sg[c] = sqrtf(G[0][c] * G[0][c] + G[1][c] * G[1][c] + G[2][c] * G[2][c]);
    auto swapcols = [&](int a, int c) {
        float t = sg[a]; sg[a] = sg[c]; sg[c] = t;
        for (int r = 0; r < 3; ++r) {
            t = G[r][a]; G[r][a] = G[r][c]; G[r][c] = t;
            t = V[r][a]; V[r][a] = V[r][c]; V[r][c] = t;
        }
    };
    if (sg[0] < sg[1]) swapcols(0, 1);
    if (sg[1] < sg[2]) swapcols(1, 2);
    if (sg[0] < sg[1]) swapcols(0, 1);

    float U[3][3];
    for (int c = 0; c < 3; ++c) {
        const float inv = (sg[c] > 1e-12f) ? (1.0f / sg[c]) : 0.0f;
        for (int r = 0; r < 3; ++r) U[r][c] = G[r][c] * inv;
    }
    if (sg[2] <= 1e-12f) {  // rank-deficient: complete U with cross product
        float ux = U[1][0] * U[2][1] - U[2][0] * U[1][1];
        float uy = U[2][0] * U[0][1] - U[0][0] * U[2][1];
        float uz = U[0][0] * U[1][1] - U[1][0] * U[0][1];
        const float nrm = sqrtf(ux * ux + uy * uy + uz * uz);
        if (nrm > 1e-20f) { U[0][2] = ux / nrm; U[1][2] = uy / nrm; U[2][2] = uz / nrm; }
    }

    const float d = det3x3(V) * det3x3(U);     // det(V @ U^T)
    float R[3][3];
    for (int a = 0; a < 3; ++a)
        for (int c = 0; c < 3; ++c)
            R[a][c] = V[a][0] * U[c][0] + V[a][1] * U[c][1] + d * V[a][2] * U[c][2];
    float tvv[3];
    for (int a = 0; a < 3; ++a)
        tvv[a] = ct[a] - (R[a][0] * cs[0] + R[a][1] * cs[1] + R[a][2] * cs[2]);

    float* o = out + b * 16;
    for (int r = 0; r < 3; ++r) {
        o[r * 4 + 0] = R[r][0]; o[r * 4 + 1] = R[r][1]; o[r * 4 + 2] = R[r][2];
        o[r * 4 + 3] = tvv[r];
    }
    o[12] = 0.0f; o[13] = 0.0f; o[14] = 0.0f; o[15] = 1.0f;
}

// ---------------------------------------------------------------------------
// Host driver
// ---------------------------------------------------------------------------
extern "C" void kernel_launch(void* const* d_in, const int* in_sizes, int n_in,
                              void* d_out, int out_size, void* d_ws, size_t ws_size,
                              hipStream_t stream) {
    (void)in_sizes; (void)n_in; (void)out_size; (void)ws_size;
    // input order (dict insertion order): pts1, pts2, then per conv {g,b,m,v,W}
    // enc0..3, dec0..2, then head_W, head_b
    static const int CIN[8]  = {64, 64, 128, 256, 2048, 512, 256, 128};
    static const int COUT[8] = {64, 128, 256, 1024, 512, 256, 128, 16};
    static const int WIDX[8] = {6, 11, 16, 21, 26, 31, 36, 37};
    static const int GIDX[7] = {2, 7, 12, 17, 22, 27, 32};

    char* ws = (char*)d_ws;
    size_t off = 0;
    auto alloc = [&](size_t bytes) -> char* {
        off = (off + 255) & ~(size_t)255;
        char* p = ws + off;
        off += bytes;
        return p;
    };

    bf16* wbf[8]; float* aP[8]; float* cP[8];
    for (int l = 0; l < 8; ++l) wbf[l] = (bf16*)alloc((size_t)CIN[l] * COUT[l] * sizeof(bf16));
    for (int l = 0; l < 8; ++l) {
        aP[l] = (float*)alloc(COUT[l] * sizeof(float));
        cP[l] = (float*)alloc(COUT[l] * sizeof(float));
    }
    int*   idxb   = (int*)  alloc((size_t)NPTS * KNN * sizeof(int));
    bf16*  Xb     = (bf16*) alloc((size_t)2048 * NPTS * sizeof(bf16));
    bf16*  Yb     = (bf16*) alloc((size_t)1024 * NPTS * sizeof(bf16));
    bf16*  glob   = (bf16*) alloc(1024 * sizeof(bf16));
    float* logits = (float*)alloc((size_t)NJ * NPTS * sizeof(float));
    float* gamma  = (float*)alloc((size_t)2 * NB * NJ * NPTS * sizeof(float));
    float* piB    = (float*)alloc((size_t)2 * NB * NJ * sizeof(float));
    float* muB    = (float*)alloc((size_t)2 * NB * NJ * 3 * sizeof(float));
    float* sigB   = (float*)alloc((size_t)2 * NB * NJ * sizeof(float));

    // one-time param prep
    for (int l = 0; l < 8; ++l) {
        const int n = CIN[l] * COUT[l];
        k_convert_w<<<(n + 255) / 256, 256, 0, stream>>>((const float*)d_in[WIDX[l]], wbf[l], n);
    }
    for (int l = 0; l < 7; ++l) {
        k_fold_bn<<<(COUT[l] + 255) / 256, 256, 0, stream>>>(
            (const float*)d_in[GIDX[l]], (const float*)d_in[GIDX[l] + 1],
            (const float*)d_in[GIDX[l] + 2], (const float*)d_in[GIDX[l] + 3],
            aP[l], cP[l], COUT[l]);
    }
    k_fill_head<<<1, 32, 0, stream>>>((const float*)d_in[38], aP[7], cP[7]);

    const int PC = NPTS / 256;   // grid.y: 256 columns per block
    for (int cl = 0; cl < 2; ++cl) {
        const float* ptsAll = (const float*)d_in[cl];
        for (int b = 0; b < NB; ++b) {
            const float* pts = ptsAll + (size_t)b * NPTS * 3;
            k_knn<<<NPTS / 256, 256, 0, stream>>>(pts, idxb);
            k_rri<<<NPTS * KNN / 128, 128, 0, stream>>>(pts, idxb, Yb);

            k_gemm_t<64, 64, 1, 0><<<dim3(4, PC), 128, 0, stream>>>(
                wbf[0], aP[0], cP[0], Yb, Xb, nullptr);            // 64   -> 64
            k_gemm_t<64, 128, 1, 0><<<dim3(8, PC), 128, 0, stream>>>(
                wbf[1], aP[1], cP[1], Xb, Yb, nullptr);            // 64   -> 128
            k_gemm_t<128, 256, 1, 0><<<dim3(16, PC), 128, 0, stream>>>(
                wbf[2], aP[2], cP[2], Yb, Xb, nullptr);            // 128  -> 256
            k_gemm_t<256, 1024, 1, 0><<<dim3(64, PC), 128, 0, stream>>>(
                wbf[3], aP[3], cP[3], Xb, Yb, nullptr);            // 256  -> 1024
            k_maxpool<<<1024, 256, 0, stream>>>(Yb, glob);
            k_concat<<<(2048 * NPTS) / 256, 256, 0, stream>>>(Yb, glob, Xb);  // -> 2048
            k_gemm_t<2048, 512, 1, 0><<<dim3(32, PC), 128, 0, stream>>>(
                wbf[4], aP[4], cP[4], Xb, Yb, nullptr);            // 2048 -> 512
            k_gemm_t<512, 256, 1, 0><<<dim3(16, PC), 128, 0, stream>>>(
                wbf[5], aP[5], cP[5], Yb, Xb, nullptr);            // 512  -> 256
            k_gemm_t<256, 128, 1, 0><<<dim3(8, PC), 128, 0, stream>>>(
                wbf[6], aP[6], cP[6], Xb, Yb, nullptr);            // 256  -> 128
            k_gemm_t<128, 16, 0, 1><<<dim3(1, PC), 128, 0, stream>>>(
                wbf[7], aP[7], cP[7], Yb, nullptr, logits);        // head -> f32 logits

            float* gcb = gamma + ((size_t)(cl * NB + b)) * NJ * NPTS;
            k_softmax<<<NPTS / 256, 256, 0, stream>>>(logits, gcb);
            k_gmm<<<1, 256, 0, stream>>>(gcb, pts,
                                         piB  + (cl * NB + b) * NJ,
                                         muB  + (size_t)(cl * NB + b) * NJ * 3,
                                         sigB + (cl * NB + b) * NJ);
        }
    }
    k_register<<<1, 16, 0, stream>>>(piB, muB, muB + (size_t)NB * NJ * 3,
                                     sigB + NB * NJ, (float*)d_out);
}
